// GRUGNNEncoder_48859547959739
// MI455X (gfx1250) — compile-verified, matrix operand order checked by reading, not run on hardware
//
#include <hip/hip_runtime.h>
#include <hip/hip_bf16.h>

// ---------------- problem constants ----------------
#define NN     1024          // nodes
#define LL     128           // graph timesteps
#define EE     16384         // edges per timestep
#define INF_   8             // input features
#define HH     64            // hidden
#define LP     129           // sequence length (L+1)
#define DI_    256           // mamba inner dim
#define DS_    10            // state dim
#define DR_    4             // dt rank
#define DC_    4             // conv width
#define NM_    7             // mixture count
#define MROWS  (NN*LP)       // 132096 GEMM rows (multiple of 16)

typedef __attribute__((ext_vector_type(16))) _Float16 v16h;
typedef __attribute__((ext_vector_type(8)))  float    v8f;

// ============================================================
// WMMA GEMM: C(MxN) = A(MxK) @ B(KxN), f32 storage, f16 math,
// f32 accumulate via v_wmma_f32_16x16x32_f16. One wave per
// 16x16 C tile. M,K must be multiples of 16/32; N may be ragged.
// Ragged columns handled by clamped addresses + f16 zero-mask so
// EXEC stays convergent and all loads are unconditional.
// ============================================================
__global__ void __launch_bounds__(128)
wmma_gemm_kernel(const float* __restrict__ A, int lda,
                 const float* __restrict__ B, int ldb,
                 float*       __restrict__ C, int ldc,
                 int M, int K, int Ncols)
{
    const int lane = threadIdx.x & 31;
    const int wave = threadIdx.x >> 5;
    const int row0 = blockIdx.x * 16;
    const int col0 = (blockIdx.y * 4 + wave) * 16;
    if (col0 >= Ncols) return;          // whole wave exits: EXEC stays all-1s for WMMA

    const int mr = lane & 15;           // A row / B,C,D column within tile
    const int kg = lane >> 4;           // K-half group (ISA 16-bit operand layout)

    // B column for this lane: clamp address in-bounds, zero OOB via mask.
    const int      bc    = col0 + mr;
    const int      bcc   = (bc < Ncols) ? bc : (Ncols - 1);
    const _Float16 bmask = (bc < Ncols) ? (_Float16)1.0f : (_Float16)0.0f;
    const float* __restrict__ Bcol = B + bcc;      // 32-bit offsets below

    const float* __restrict__ Arow = A + (size_t)(row0 + mr) * lda;

    v8f acc = {};
    for (int k0 = 0; k0 < K; k0 += 32) {
        // ---- A tile 16x32 (row-major src) -> ISA A layout ----
        const float* ap = Arow + k0;
        if (k0 + 32 < K) __builtin_prefetch(ap + 32, 0, 0);   // global_prefetch_b8
        v16h a;
        #pragma unroll
        for (int v = 0; v < 8; ++v) {
            const int kbase = ((v & 4) ? 16 : 0) + (v & 3) * 2 + kg * 8;
            const float2 f2 = *(const float2*)(ap + kbase);
            a[2*v]   = (_Float16)f2.x;
            a[2*v+1] = (_Float16)f2.y;
        }
        // ---- B tile 32x16 (row-major src, K x N), unconditional loads ----
        v16h b;
        #pragma unroll
        for (int j = 0; j < 16; ++j) {
            const int v = j >> 1, e = j & 1;
            const int k = ((v & 4) ? 16 : 0) + (v & 3) * 2 + e + kg * 8;
            const int boff = (k0 + k) * ldb;       // 32-bit address math
            b[j] = (_Float16)Bcol[boff] * bmask;
        }
        acc = __builtin_amdgcn_wmma_f32_16x16x32_f16(false, a, false, b,
                                                     (short)0, acc, false, false);
    }
    // ---- D store: VGPR r -> row r + 8*kg, col = mr ----
    const int cc = col0 + mr;
    if (cc < Ncols) {
        #pragma unroll
        for (int r = 0; r < 8; ++r)
            C[(size_t)(row0 + r + 8 * kg) * ldc + cc] = acc[r];
    }
}

// ============================================================
// GNN stage
// ============================================================
__global__ void zero_kernel(float* __restrict__ p, int n)
{
    int i = blockIdx.x * blockDim.x + threadIdx.x;
    if (i < n) p[i] = 0.f;
}

// msg = x[src,l,:] * w ; agg[l,dst,:] += msg   (atomic segment-sum)
__global__ void scatter_kernel(const float* __restrict__ x,
                               const int*   __restrict__ ei,
                               const float* __restrict__ ef,
                               float*       __restrict__ agg)
{
    int idx = blockIdx.x * blockDim.x + threadIdx.x;   // LL*EE threads
    if (idx >= LL * EE) return;
    const int e = idx % EE, l = idx / EE;
    const int src = ei[((size_t)l * 2    ) * EE + e];
    const int dst = ei[((size_t)l * 2 + 1) * EE + e];
    const float w = ef[(size_t)l * EE + e];
    const float* xs = x   + ((size_t)src * LL + l) * INF_;
    float*       ag = agg + ((size_t)l  * NN + dst) * INF_;
    #pragma unroll
    for (int i = 0; i < INF_; ++i) atomicAdd(&ag[i], xs[i] * w);
}

// gates[l,n,k] = x[n,l,:]@w_root[:,k] + agg[l,n,:]@w_nei[:,k] + gnn_b[k] + cell_bias[k]
// split into seq1 (k<64, at t=l+1) / seq2 (k>=64)
__global__ void gates_kernel(const float* __restrict__ x,
                             const float* __restrict__ agg,
                             const float* __restrict__ w_root,
                             const float* __restrict__ w_nei,
                             const float* __restrict__ gb,
                             const float* __restrict__ cb,
                             float* __restrict__ seq1,
                             float* __restrict__ seq2)
{
    int idx = blockIdx.x * blockDim.x + threadIdx.x;   // LL*NN*128 threads
    if (idx >= LL * NN * 128) return;
    const int k = idx & 127;
    const int n = (idx >> 7) & (NN - 1);
    const int l = idx >> 17;
    float s = gb[k] + cb[k];
    const float* xr = x   + ((size_t)n * LL + l) * INF_;
    const float* ar = agg + ((size_t)l * NN + n) * INF_;
    #pragma unroll
    for (int i = 0; i < INF_; ++i)
        s += xr[i] * w_root[i * 128 + k] + ar[i] * w_nei[i * 128 + k];
    if (k < HH) seq1[((size_t)n * LP + l + 1) * HH + k]        = s;
    else        seq2[((size_t)n * LP + l + 1) * HH + (k - HH)] = s;
}

__global__ void init_seq_kernel(const float* __restrict__ h0,
                                float* __restrict__ s1, float* __restrict__ s2)
{
    int idx = blockIdx.x * blockDim.x + threadIdx.x;   // NN*HH
    if (idx >= NN * HH) return;
    const int n = idx >> 6, h = idx & 63;
    s1[(size_t)n * LP * HH + h] = h0[h];
    s2[(size_t)n * LP * HH + h] = h0[h];
}

// ============================================================
// Mamba stage kernels
// ============================================================
// depthwise causal conv (DC=4) + bias + SiLU ; reads xi half of xz (stride 512)
__global__ void conv_silu_kernel(const float* __restrict__ xz,
                                 const float* __restrict__ cw,
                                 const float* __restrict__ cb,
                                 float* __restrict__ xc)
{
    int idx = blockIdx.x * blockDim.x + threadIdx.x;   // MROWS*DI_ threads
    if (idx >= MROWS * DI_) return;
    const int c = idx & 255;
    const int t = (idx >> 8) % LP;
    const int n = idx / (LP * DI_);
    const float* base = xz + ((size_t)n * LP + t) * 512 + c;
    float s = cb[c];
    #pragma unroll
    for (int j = 0; j < DC_; ++j) {
        const int tt = t + j - (DC_ - 1);
        if (tt >= 0) s += base[(tt - t) * 512] * cw[c * DC_ + j];
    }
    const float sig = 1.f / (1.f + __expf(-s));
    xc[idx] = s * sig;
}

// selective scan: one block per node, one thread per channel.
// dt projection + softplus computed on the fly from LDS-staged dbl row.
// y_full written in-place over dead xi region of xz (stride 512).
__global__ void __launch_bounds__(256)
scan_kernel(const float* __restrict__ dbl,
            const float* __restrict__ xc,
            float*       __restrict__ xzbuf,   // z @ +256, y out @ +0
            const float* __restrict__ dt_w,
            const float* __restrict__ dt_b,
            const float* __restrict__ A_log,
            const float* __restrict__ Dv)
{
    __shared__ float sd[DR_ + 2 * DS_];            // 24 floats: dt_raw | B | C
    const int n = blockIdx.x;
    const int c = threadIdx.x;
    float As[DS_], h[DS_];
    #pragma unroll
    for (int s = 0; s < DS_; ++s) { As[s] = -__expf(A_log[c * DS_ + s]); h[s] = 0.f; }
    const float w0 = dt_w[c], w1 = dt_w[DI_ + c], w2 = dt_w[2*DI_ + c], w3 = dt_w[3*DI_ + c];
    const float bdt = dt_b[c], Dc = Dv[c];

    for (int t = 0; t < LP; ++t) {
        const size_t row = (size_t)n * LP + t;
        if (threadIdx.x < 24) sd[threadIdx.x] = dbl[row * 24 + threadIdx.x];
        __syncthreads();
        const float dtr = sd[0]*w0 + sd[1]*w1 + sd[2]*w2 + sd[3]*w3 + bdt;
        const float dt  = (dtr > 20.f) ? dtr : log1pf(__expf(dtr));
        const float xv  = xc[row * DI_ + c];
        const float dx  = dt * xv;
        float y = 0.f;
        #pragma unroll
        for (int s = 0; s < DS_; ++s) {
            h[s] = h[s] * __expf(dt * As[s]) + dx * sd[DR_ + s];
            y   += h[s] * sd[DR_ + DS_ + s];
        }
        const float zv  = xzbuf[row * 512 + DI_ + c];
        const float sig = 1.f / (1.f + __expf(-zv));
        xzbuf[row * 512 + c] = (y + xv * Dc) * (zv * sig);
        __syncthreads();
    }
}

// layernorm(64) -> tanh ; optionally accumulate into out
__global__ void ln_tanh_kernel(const float* __restrict__ X,
                               const float* __restrict__ g,
                               const float* __restrict__ b,
                               float* __restrict__ out, int accumulate, int rows)
{
    int r = blockIdx.x * blockDim.x + threadIdx.x;
    if (r >= rows) return;
    const float* xr = X + (size_t)r * HH;
    float m = 0.f;
    for (int i = 0; i < HH; ++i) m += xr[i];
    m *= (1.f / HH);
    float v = 0.f;
    for (int i = 0; i < HH; ++i) { const float d = xr[i] - m; v += d * d; }
    v *= (1.f / HH);
    const float inv = rsqrtf(v + 1e-5f);
    float* o = out + (size_t)r * HH;
    for (int i = 0; i < HH; ++i) {
        const float t = tanhf((xr[i] - m) * inv * g[i] + b[i]);
        o[i] = accumulate ? (o[i] + t) : t;
    }
}

// m4 input: flip(out1, t) * out2
__global__ void build_m4_kernel(const float* __restrict__ s1,
                                const float* __restrict__ s2,
                                float* __restrict__ o)
{
    int idx = blockIdx.x * blockDim.x + threadIdx.x;   // MROWS*HH
    if (idx >= MROWS * HH) return;
    const int h = idx & 63;
    const int t = (idx >> 6) % LP;
    const int n = idx / (LP * HH);
    o[idx] = s1[((size_t)n * LP + (LP - 1 - t)) * HH + h] * s2[idx];
}

// mixture_w = tanh(out[:, -1]) @ mix_w + mix_b
__global__ void mixture_kernel(const float* __restrict__ outbuf,
                               const float* __restrict__ mw,
                               const float* __restrict__ mb,
                               float* __restrict__ mix)
{
    int idx = blockIdx.x * blockDim.x + threadIdx.x;   // NN*NM_
    if (idx >= NN * NM_) return;
    const int n = idx / NM_, j = idx % NM_;
    const float* row = outbuf + ((size_t)n * LP + (LP - 1)) * HH;
    float s = mb[j];
    for (int h = 0; h < HH; ++h) s += tanhf(row[h]) * mw[h * NM_ + j];
    mix[idx] = s;
}

// ============================================================
// Host-side mamba block orchestration (5 launches, all WMMA GEMMs)
// ============================================================
static void run_mamba(const float* X, void* const* P,
                      float* xz, float* xc, float* dbl, float* Yout,
                      hipStream_t stream)
{
    const float* in_w   = (const float*)P[0];
    const float* conv_w = (const float*)P[1];
    const float* conv_b = (const float*)P[2];
    const float* x_w    = (const float*)P[3];
    const float* dt_w   = (const float*)P[4];
    const float* dt_b   = (const float*)P[5];
    const float* A_log  = (const float*)P[6];
    const float* Dv     = (const float*)P[7];
    const float* out_w  = (const float*)P[8];
    const int M = MROWS;

    // xz = X @ in_w   (132096 x 64 x 512)
    wmma_gemm_kernel<<<dim3(M/16, 8), 128, 0, stream>>>(X, HH, in_w, 2*DI_, xz, 2*DI_, M, HH, 2*DI_);
    // xc = silu(causal_dwconv(xi) + b)
    conv_silu_kernel<<<(M*DI_ + 255)/256, 256, 0, stream>>>(xz, conv_w, conv_b, xc);
    // dbl = xc @ x_w  (132096 x 256 x 24)
    wmma_gemm_kernel<<<dim3(M/16, 1), 128, 0, stream>>>(xc, DI_, x_w, DR_+2*DS_, dbl, DR_+2*DS_, M, DI_, DR_+2*DS_);
    // selective scan (dt proj fused), y written over xi region of xz
    scan_kernel<<<NN, DI_, 0, stream>>>(dbl, xc, xz, dt_w, dt_b, A_log, Dv);
    // Yout = y @ out_w (132096 x 256 x 64)
    wmma_gemm_kernel<<<dim3(M/16, 1), 128, 0, stream>>>(xz, 2*DI_, out_w, HH, Yout, HH, M, DI_, HH);
}

// ============================================================
extern "C" void kernel_launch(void* const* d_in, const int* in_sizes, int n_in,
                              void* d_out, int out_size, void* d_ws, size_t ws_size,
                              hipStream_t stream)
{
    (void)in_sizes; (void)n_in; (void)out_size; (void)ws_size;
    const float* x      = (const float*)d_in[0];
    const int*   ei     = (const int*)  d_in[1];
    const float* ef     = (const float*)d_in[2];
    const float* w_root = (const float*)d_in[3];
    const float* w_nei  = (const float*)d_in[4];
    const float* gnn_b  = (const float*)d_in[5];
    const float* cell_b = (const float*)d_in[6];
    const float* h0     = (const float*)d_in[7];
    void* const* m3P    = d_in + 8;
    void* const* m4P    = d_in + 17;
    void* const* m5P    = d_in + 26;
    const float* ln3_g  = (const float*)d_in[35];
    const float* ln3_b  = (const float*)d_in[36];
    const float* ln4_g  = (const float*)d_in[37];
    const float* ln4_b  = (const float*)d_in[38];
    const float* mix_w  = (const float*)d_in[39];
    const float* mix_b  = (const float*)d_in[40];
    float* out = (float*)d_out;

    // ---- workspace carve (256B aligned) ----
    char* ws = (char*)d_ws;
    size_t off = 0;
    auto carve = [&](size_t nfloat) -> float* {
        float* p = (float*)(ws + off);
        off = (off + nfloat * sizeof(float) + 255) & ~(size_t)255;
        return p;
    };
    float* agg  = carve((size_t)LL * NN * INF_);     //   4 MB
    float* seq1 = carve((size_t)MROWS * HH);         //  34 MB
    float* seq2 = carve((size_t)MROWS * HH);
    float* tmp  = carve((size_t)MROWS * HH);
    float* acc  = carve((size_t)MROWS * HH);
    float* mo   = carve((size_t)MROWS * HH);
    float* xz   = carve((size_t)MROWS * 2 * DI_);    // 270 MB (xi|z, y overwrites xi)
    float* xc   = carve((size_t)MROWS * DI_);        // 135 MB
    float* dbl  = carve((size_t)MROWS * (DR_ + 2*DS_)); // 13 MB

    // ---- GNN: segment-sum + gate projection, build sequences ----
    zero_kernel<<<(LL*NN*INF_ + 255)/256, 256, 0, stream>>>(agg, LL*NN*INF_);
    scatter_kernel<<<(LL*EE + 255)/256, 256, 0, stream>>>(x, ei, ef, agg);
    init_seq_kernel<<<(NN*HH + 255)/256, 256, 0, stream>>>(h0, seq1, seq2);
    gates_kernel<<<(LL*NN*128 + 255)/256, 256, 0, stream>>>(x, agg, w_root, w_nei,
                                                            gnn_b, cell_b, seq1, seq2);

    // ---- o3 = tanh(ln3(mamba3(out1))) ----
    run_mamba(seq1, m3P, xz, xc, dbl, mo, stream);
    ln_tanh_kernel<<<(MROWS + 255)/256, 256, 0, stream>>>(mo, ln3_g, ln3_b, acc, 0, MROWS);

    // ---- o4 = tanh(ln4(mamba4(flip(out1) * out2))) ; acc = o3 + o4 ----
    build_m4_kernel<<<(MROWS*HH + 255)/256, 256, 0, stream>>>(seq1, seq2, tmp);
    run_mamba(tmp, m4P, xz, xc, dbl, mo, stream);
    ln_tanh_kernel<<<(MROWS + 255)/256, 256, 0, stream>>>(mo, ln4_g, ln4_b, acc, 1, MROWS);

    // ---- out = mamba5(o3 + o4) straight into d_out ----
    run_mamba(acc, m5P, xz, xc, dbl, out, stream);

    // ---- mixture head ----
    mixture_kernel<<<(NN*NM_ + 255)/256, 256, 0, stream>>>(out, mix_w, mix_b,
                                                           out + (size_t)MROWS * HH);
}
